// MultiHeadAttention_27659589386889
// MI455X (gfx1250) — matrix operand order repaired and checked by reading, not compiled, not run on hardware
//
#include <hip/hip_runtime.h>
#include <hip/hip_bf16.h>
#include <math.h>

// ---------------- CDNA5 (gfx1250) MHA forward ----------------
// wave32, WMMA f32_16x16x32_f16. Fragment operands loaded as paired 16-byte
// (b128) loads exploiting contiguous-K runs in the CDNA5 16-bit layouts:
//   A-frag: lane holds K = [8g..8g+7] ++ [16+8g..23+8g]   (g = lane>>4)
//   B-frag: lane holds K = [8g..8g+7] ++ [16+8g..23+8g]
// Tile staging uses GLOBAL_LOAD_ASYNC_TO_LDS_B128 (ASYNCcnt) when the
// toolchain exposes the builtin, else synchronous b128 copies.
// Kernels: 0a) f32->f16 converts (x + 4 weights)  0b) RoPE cos/sin table
//          1) QKV proj + RoPE  2) causal flash-attn  3) out projection -> f32

typedef __attribute__((ext_vector_type(16))) _Float16 v16h;
typedef __attribute__((ext_vector_type(8)))  _Float16 v8h;
typedef __attribute__((ext_vector_type(4)))  _Float16 v4h;
typedef __attribute__((ext_vector_type(8)))  float    v8f;
typedef __attribute__((ext_vector_type(4)))  float    v4f;

#define SEQL   2048
#define DMODEL 1024
#define NHEADS 16
#define DKH    64

#if defined(__has_builtin)
#if __has_builtin(__builtin_amdgcn_global_load_async_to_lds_b128) && \
    __has_builtin(__builtin_amdgcn_s_wait_asynccnt)
#define HAVE_GFX1250_ASYNC 1
#endif
#endif

#ifdef HAVE_GFX1250_ASYNC
typedef int v4i_vs __attribute__((vector_size(4 * sizeof(int))));
typedef __attribute__((address_space(1))) v4i_vs* g_b128_ptr;   // global
typedef __attribute__((address_space(3))) v4i_vs* l_b128_ptr;   // LDS
#endif

// 16-byte global -> LDS copy (async DMA path when available).
__device__ __forceinline__ void copy16_g2l(_Float16* lds, const _Float16* g) {
#ifdef HAVE_GFX1250_ASYNC
    __builtin_amdgcn_global_load_async_to_lds_b128(
        (g_b128_ptr)(g), (l_b128_ptr)(lds), 0, 0);
#else
    *(v8h*)lds = *(const v8h*)g;
#endif
}
__device__ __forceinline__ void async_fence() {
#ifdef HAVE_GFX1250_ASYNC
    __builtin_amdgcn_s_wait_asynccnt(0);
#endif
}

__device__ __forceinline__ v16h concat8(v8h lo, v8h hi) {
    return __builtin_shufflevector(lo, hi, 0, 1, 2, 3, 4, 5, 6, 7,
                                           8, 9, 10, 11, 12, 13, 14, 15);
}
// A fragment: p = row base at the 32-wide K-chunk origin (16B aligned).
__device__ __forceinline__ v16h load_afrag(const _Float16* p, int grp) {
    v8h lo = *(const v8h*)(p + 8 * grp);
    v8h hi = *(const v8h*)(p + 16 + 8 * grp);
    return concat8(lo, hi);
}
// B fragment: p = this lane's column K-run origin (16B aligned).
__device__ __forceinline__ v16h load_bfrag(const _Float16* p, int grp) {
    v8h lo = *(const v8h*)(p + 8 * grp);
    v8h hi = *(const v8h*)(p + 16 + 8 * grp);
    return concat8(lo, hi);
}

__device__ __forceinline__ float red_max16(float v) {
    v = fmaxf(v, __shfl_xor(v, 1, 32));
    v = fmaxf(v, __shfl_xor(v, 2, 32));
    v = fmaxf(v, __shfl_xor(v, 4, 32));
    v = fmaxf(v, __shfl_xor(v, 8, 32));
    return v;
}
__device__ __forceinline__ float red_sum16(float v) {
    v += __shfl_xor(v, 1, 32);
    v += __shfl_xor(v, 2, 32);
    v += __shfl_xor(v, 4, 32);
    v += __shfl_xor(v, 8, 32);
    return v;
}

// -------- Kernel 0a: f32 -> f16 (coalesced, 4 elems/thread) --------
__global__ void __launch_bounds__(256)
cvt_f16_kernel(const float* __restrict__ x, _Float16* __restrict__ xh) {
    const size_t i = ((size_t)blockIdx.x * 256 + threadIdx.x) * 4;
    v4f v = *(const v4f*)(x + i);
    v4h hv = {(_Float16)v[0], (_Float16)v[1], (_Float16)v[2], (_Float16)v[3]};
    *(v4h*)(xh + i) = hv;
}

// -------- Kernel 0b: RoPE table tab[pos][ip] = (cos, sin) --------
// freq = 10000^(-2*ip/64) = 10^(-ip/8)
__global__ void __launch_bounds__(256)
rope_tab_kernel(float2* __restrict__ tab) {
    const int idx = blockIdx.x * 256 + threadIdx.x;   // 2048*32 total
    const int pos = idx >> 5, ip = idx & 31;
    float freq = __exp10f(-(float)ip * 0.125f);
    float ang = (float)pos * freq;
    float sn, cs;
    __sincosf(ang, &sn, &cs);
    tab[idx] = make_float2(cs, sn);
}

// -------- Kernel 1: projection + optional RoPE --------
// grid (M/64, N/64, 3), block 128 (4 waves). z: 0=Q(rope),1=K(rope),2=V.
__global__ void __launch_bounds__(128)
proj_rope_kernel(const _Float16* __restrict__ xh,
                 const _Float16* __restrict__ Wqh,
                 const _Float16* __restrict__ Wkh,
                 const _Float16* __restrict__ Wvh,
                 const int*   __restrict__ pos_ids,
                 const float2* __restrict__ rope_tab,
                 _Float16* __restrict__ qb, _Float16* __restrict__ kb,
                 _Float16* __restrict__ vb)
{
    // col-major W^T tile: btile[n][k], stride 40 f16 (80B, 16B aligned)
    __shared__ __align__(16) _Float16 btile[64][40];

    const int which = blockIdx.z;
    const _Float16* __restrict__ W =
        (which == 0) ? Wqh : ((which == 1) ? Wkh : Wvh);
    _Float16* __restrict__ ob = (which == 0) ? qb : ((which == 1) ? kb : vb);
    const bool do_rope = (which < 2);

    const int tid  = threadIdx.x;
    const int lane = tid & 31;
    const int w    = tid >> 5;
    const int grp  = lane >> 4;
    const int colc = lane & 15;
    const int m0   = blockIdx.x * 64;
    const int n0   = blockIdx.y * 64;
    const int K    = DMODEL;
    const int arow = m0 + 16 * w + (lane & 15);
    const _Float16* __restrict__ xrow = xh + (size_t)arow * K;

    v8f c0 = {}, c1 = {}, c2 = {}, c3 = {};

    for (int k0 = 0; k0 < K; k0 += 32) {
        __syncthreads();
        // btile[nn][kk8..+7] = W[n0+nn][k0+kk8..] (64 rows x 4 b128 chunks)
        for (int idx = tid; idx < 256; idx += 128) {
            int kk8 = (idx & 3) * 8, nn = idx >> 2;
            copy16_g2l(&btile[nn][kk8], W + (size_t)(n0 + nn) * K + (k0 + kk8));
        }
        async_fence();
        __syncthreads();

        __builtin_prefetch(xrow + k0 + 32, 0, 3);   // global_prefetch_b8 (near)
        v16h a  = load_afrag(xrow + k0, grp);
        v16h b0 = load_bfrag(&btile[colc][0], grp);
        v16h b1 = load_bfrag(&btile[16 + colc][0], grp);
        v16h b2 = load_bfrag(&btile[32 + colc][0], grp);
        v16h b3 = load_bfrag(&btile[48 + colc][0], grp);
        c0 = __builtin_amdgcn_wmma_f32_16x16x32_f16(false, a, false, b0, (short)0, c0, false, false);
        c1 = __builtin_amdgcn_wmma_f32_16x16x32_f16(false, a, false, b1, (short)0, c1, false, false);
        c2 = __builtin_amdgcn_wmma_f32_16x16x32_f16(false, a, false, b2, (short)0, c2, false, false);
        c3 = __builtin_amdgcn_wmma_f32_16x16x32_f16(false, a, false, b3, (short)0, c3, false, false);
    }

    // epilogue: optional RoPE via table (partner value lives in lane^1)
    v8f cc[4] = {c0, c1, c2, c3};
    #pragma unroll
    for (int i = 0; i < 8; ++i) {
        const int row  = i + 8 * grp;
        const int mrow = m0 + 16 * w + row;
        const int bb   = mrow >> 11;           // / SEQL
        const int ss   = mrow & (SEQL - 1);
        const int pos  = do_rope ? pos_ids[bb * SEQL + ss] : 0;
        #pragma unroll
        for (int f = 0; f < 4; ++f) {
            float val = cc[f][i];
            float partner = __shfl_xor(val, 1, 32);
            const int e  = n0 + 16 * f + colc;
            const int dk = e & (DKH - 1);
            if (do_rope) {
                float2 cssn = rope_tab[pos * 32 + (dk >> 1)];
                val = (dk & 1) ? (partner * cssn.y + val * cssn.x)
                               : (val * cssn.x - partner * cssn.y);
            }
            const int hh = e >> 6;
            ob[(((size_t)(bb * NHEADS + hh) * SEQL + ss) << 6) + dk] = (_Float16)val;
        }
    }
}

// -------- Kernel 2: causal flash-attention --------
// grid (S/64, B*H), block 128 (4 waves, each owns 16 q rows).
// K tile staged via async-to-LDS; V transposed via scatter.
__global__ void __launch_bounds__(128)
attn_kernel(const _Float16* __restrict__ qb, const _Float16* __restrict__ kb,
            const _Float16* __restrict__ vb, _Float16* __restrict__ ob)
{
    __shared__ __align__(16) _Float16 ktile[32][72];      // [kc][d]   (144B rows)
    __shared__ __align__(16) _Float16 vtile[64][40];      // [d][krel] (80B rows)
    __shared__ __align__(16) _Float16 pbuf[4][16][40];    // per-wave P

    const int tid  = threadIdx.x;
    const int lane = tid & 31;
    const int w    = tid >> 5;
    const int grp  = lane >> 4;
    const int colc = lane & 15;
    const int q0   = blockIdx.x * 64;
    const int bh   = blockIdx.y;
    const int b    = bh >> 4;
    const size_t base = ((size_t)bh * SEQL) << 6;   // (b*H+h)*S*64

    // resident Q A-fragments (d_k = 64 -> two 32-chunks)
    const int qrow = q0 + 16 * w + (lane & 15);
    const _Float16* __restrict__ qp = qb + base + ((size_t)qrow << 6);
    v16h aq0 = load_afrag(qp, grp);
    v16h aq1 = load_afrag(qp + 32, grp);

    v8f o0 = {}, o1 = {}, o2 = {}, o3 = {};
    float mrow[8], lrow[8];
    #pragma unroll
    for (int i = 0; i < 8; ++i) { mrow[i] = -__builtin_inff(); lrow[i] = 0.0f; }

    const int qlast = q0 + 16 * w + 15;    // this wave's causal limit
    const float scale = 0.125f;            // 1/sqrt(64)

    for (int k0 = 0; k0 <= q0 + 63; k0 += 32) {
        __syncthreads();
        // stage K (async b128 g->LDS) and V (transposed scatter)
        for (int idx = tid; idx < 256; idx += 128) {
            int d8 = (idx & 7) * 8, kr = idx >> 3;
            const size_t g = base + (((size_t)(k0 + kr)) << 6) + d8;
            copy16_g2l(&ktile[kr][d8], kb + g);
            v8h vv = *(const v8h*)(vb + g);
            #pragma unroll
            for (int j = 0; j < 8; ++j) vtile[d8 + j][kr] = vv[j];
        }
        async_fence();
        __syncthreads();

        if (k0 <= qlast) {
            // S = Q K^T, two 16-wide column fragments
            v8f sf[2];
            #pragma unroll
            for (int sfi = 0; sfi < 2; ++sfi) {
                const _Float16* kp = &ktile[16 * sfi + colc][0];
                v16h bk0 = load_bfrag(kp, grp);
                v16h bk1 = load_bfrag(kp + 32, grp);
                v8f acc = {};
                acc = __builtin_amdgcn_wmma_f32_16x16x32_f16(false, aq0, false, bk0, (short)0, acc, false, false);
                acc = __builtin_amdgcn_wmma_f32_16x16x32_f16(false, aq1, false, bk1, (short)0, acc, false, false);
                sf[sfi] = acc;
            }

            // causal mask + online softmax stats; P -> per-wave LDS (f16)
            float alpha[8];
            #pragma unroll
            for (int i = 0; i < 8; ++i) {
                const int qr = q0 + 16 * w + i + 8 * grp;
                float s0 = sf[0][i] * scale;
                float s1 = sf[1][i] * scale;
                if (k0 + colc      > qr) s0 = -__builtin_inff();
                if (k0 + 16 + colc > qr) s1 = -__builtin_inff();
                float tmax = red_max16(fmaxf(s0, s1));
                float nm = fmaxf(mrow[i], tmax);
                alpha[i] = __expf(mrow[i] - nm);
                mrow[i] = nm;
                float p0 = __expf(s0 - nm);
                float p1 = __expf(s1 - nm);
                const int row = i + 8 * grp;
                pbuf[w][row][colc]      = (_Float16)p0;
                pbuf[w][row][16 + colc] = (_Float16)p1;
                lrow[i] = lrow[i] * alpha[i] + red_sum16(p0 + p1);
            }
            #pragma unroll
            for (int i = 0; i < 8; ++i) {
                o0[i] *= alpha[i]; o1[i] *= alpha[i];
                o2[i] *= alpha[i]; o3[i] *= alpha[i];
            }

            // P as A-fragment (same-wave LDS RAW; DScnt waits by compiler)
            v16h ap = load_afrag(&pbuf[w][lane & 15][0], grp);

            // O += P V  (4 d-column fragments from transposed V tile)
            v16h bv0 = load_bfrag(&vtile[colc][0], grp);
            v16h bv1 = load_bfrag(&vtile[16 + colc][0], grp);
            v16h bv2 = load_bfrag(&vtile[32 + colc][0], grp);
            v16h bv3 = load_bfrag(&vtile[48 + colc][0], grp);
            o0 = __builtin_amdgcn_wmma_f32_16x16x32_f16(false, ap, false, bv0, (short)0, o0, false, false);
            o1 = __builtin_amdgcn_wmma_f32_16x16x32_f16(false, ap, false, bv1, (short)0, o1, false, false);
            o2 = __builtin_amdgcn_wmma_f32_16x16x32_f16(false, ap, false, bv2, (short)0, o2, false, false);
            o3 = __builtin_amdgcn_wmma_f32_16x16x32_f16(false, ap, false, bv3, (short)0, o3, false, false);
        }
    }

    // normalize and store f16 [B, S, 1024]  (e = h*64 + d)
    const int h = bh & 15;
    v8f oo[4] = {o0, o1, o2, o3};
    #pragma unroll
    for (int f = 0; f < 4; ++f) {
        #pragma unroll
        for (int i = 0; i < 8; ++i) {
            const int row = i + 8 * grp;
            const int s = q0 + 16 * w + row;
            const int e = h * DKH + 16 * f + colc;
            float val = oo[f][i] / lrow[i];
            ob[(((size_t)(b * SEQL + s)) << 10) + e] = (_Float16)val;
        }
    }
}

// -------- Kernel 3: output projection (attn @ Wo^T), f32 out --------
__global__ void __launch_bounds__(128)
oproj_kernel(const _Float16* __restrict__ A, const _Float16* __restrict__ Woh,
             float* __restrict__ out)
{
    __shared__ __align__(16) _Float16 btile[64][40];

    const int tid  = threadIdx.x;
    const int lane = tid & 31;
    const int w    = tid >> 5;
    const int grp  = lane >> 4;
    const int colc = lane & 15;
    const int m0   = blockIdx.x * 64;
    const int n0   = blockIdx.y * 64;
    const int K    = DMODEL;
    const int arow = m0 + 16 * w + (lane & 15);
    const _Float16* __restrict__ arp = A + (size_t)arow * K;

    v8f c0 = {}, c1 = {}, c2 = {}, c3 = {};

    for (int k0 = 0; k0 < K; k0 += 32) {
        __syncthreads();
        for (int idx = tid; idx < 256; idx += 128) {
            int kk8 = (idx & 3) * 8, nn = idx >> 2;
            copy16_g2l(&btile[nn][kk8], Woh + (size_t)(n0 + nn) * K + (k0 + kk8));
        }
        async_fence();
        __syncthreads();

        __builtin_prefetch(arp + k0 + 32, 0, 3);
        v16h a  = load_afrag(arp + k0, grp);
        v16h b0 = load_bfrag(&btile[colc][0], grp);
        v16h b1 = load_bfrag(&btile[16 + colc][0], grp);
        v16h b2 = load_bfrag(&btile[32 + colc][0], grp);
        v16h b3 = load_bfrag(&btile[48 + colc][0], grp);
        c0 = __builtin_amdgcn_wmma_f32_16x16x32_f16(false, a, false, b0, (short)0, c0, false, false);
        c1 = __builtin_amdgcn_wmma_f32_16x16x32_f16(false, a, false, b1, (short)0, c1, false, false);
        c2 = __builtin_amdgcn_wmma_f32_16x16x32_f16(false, a, false, b2, (short)0, c2, false, false);
        c3 = __builtin_amdgcn_wmma_f32_16x16x32_f16(false, a, false, b3, (short)0, c3, false, false);
    }

    v8f cc[4] = {c0, c1, c2, c3};
    #pragma unroll
    for (int f = 0; f < 4; ++f) {
        #pragma unroll
        for (int i = 0; i < 8; ++i) {
            const int row  = i + 8 * grp;
            const int mrow = m0 + 16 * w + row;
            const int n    = n0 + 16 * f + colc;
            out[(size_t)mrow * DMODEL + n] = cc[f][i];
        }
    }
}

extern "C" void kernel_launch(void* const* d_in, const int* in_sizes, int n_in,
                              void* d_out, int out_size, void* d_ws, size_t ws_size,
                              hipStream_t stream) {
    const float* x   = (const float*)d_in[0];
    const float* Wq  = (const float*)d_in[1];
    const float* Wk  = (const float*)d_in[2];
    const float* Wv  = (const float*)d_in[3];
    const float* Wo  = (const float*)d_in[4];
    const int*   pos = (const int*)  d_in[5];

    const size_t per  = (size_t)2 * NHEADS * SEQL * DKH;  // 4,194,304 f16 elems
    const size_t wsz  = (size_t)DMODEL * DMODEL;          // 1,048,576 per weight
    _Float16* xh  = (_Float16*)d_ws;                      // [B*S, 1024] f16
    _Float16* qb  = xh + per;
    _Float16* kb  = qb + per;
    _Float16* vb  = kb + per;
    _Float16* ab  = vb + per;                             // [B,S,1024] f16
    _Float16* wqh = ab + per;                             // f16 weights
    _Float16* wkh = wqh + wsz;
    _Float16* wvh = wkh + wsz;
    _Float16* woh = wvh + wsz;
    float2*   rt  = (float2*)(woh + wsz);                 // 2048*32 cos/sin
    // total ws use = 48 MB + 512 KB

    // 0a) f32 -> f16 conversions (x and the four weight matrices)
    cvt_f16_kernel<<<dim3((unsigned)(per / 1024)), 256, 0, stream>>>(x, xh);
    cvt_f16_kernel<<<dim3((unsigned)(wsz / 1024)), 256, 0, stream>>>(Wq, wqh);
    cvt_f16_kernel<<<dim3((unsigned)(wsz / 1024)), 256, 0, stream>>>(Wk, wkh);
    cvt_f16_kernel<<<dim3((unsigned)(wsz / 1024)), 256, 0, stream>>>(Wv, wvh);
    cvt_f16_kernel<<<dim3((unsigned)(wsz / 1024)), 256, 0, stream>>>(Wo, woh);
    // 0b) RoPE table
    rope_tab_kernel<<<dim3((SEQL * 32) / 256), 256, 0, stream>>>(rt);
    // 1) Q/K/V projections + RoPE (z: 0=Q,1=K,2=V)
    proj_rope_kernel<<<dim3(64, 16, 3), 128, 0, stream>>>(xh, wqh, wkh, wvh,
                                                          pos, rt, qb, kb, vb);
    // 2) causal flash-attention
    attn_kernel<<<dim3(SEQL / 64, 2 * NHEADS), 128, 0, stream>>>(qb, kb, vb, ab);
    // 3) output projection -> f32 d_out
    oproj_kernel<<<dim3(64, 16), 128, 0, stream>>>(ab, woh, (float*)d_out);
}
